// RNNModel_74672301408955
// MI455X (gfx1250) — compile-verified, matrix operand order checked by reading, not run on hardware
//
#include <hip/hip_runtime.h>

// ---------------------------------------------------------------------------
// Types for CDNA5 WMMA
// ---------------------------------------------------------------------------
typedef _Float16 h16;
typedef __attribute__((ext_vector_type(8)))  _Float16 v8h;
typedef __attribute__((ext_vector_type(16))) _Float16 v16h;
typedef __attribute__((ext_vector_type(8)))  float    v8f;

__device__ __forceinline__ float sgm(float x) { return 1.0f / (1.0f + __expf(-x)); }

// ---------------------------------------------------------------------------
// Problem dimensions (padded so every GEMM has N % 64 == 0 and K % 64 == 0)
// ---------------------------------------------------------------------------
constexpr int TT = 128, BB = 16, MM = TT * BB;          // 2048 rows (t*B+b)
constexpr int NHID = 1024;
constexpr int G3 = 3 * NHID;                            // 3072
constexpr int G4 = 4 * NHID;                            // 4096
constexpr int SS = 2186, SSP = 2240;                    // sememe K padded to %64
constexpr int SD = 300,  SDP = 320;                     // sem dim padded (%64)
constexpr int VV = 50257, VP = 50304;                   // vocab padded (%64)
constexpr size_t DEC_ELEMS = (size_t)MM * VV;

constexpr int GWAVES = 8;   // waves per block (256 threads)
constexpr int GNT    = 4;   // 16-wide N tiles per wave (N strip of 64)

// ---------------------------------------------------------------------------
// Generic NT GEMM:  C[M,N] (+)= A[M,K] * B[N,K]^T (+ bias)
// A, B are f16 row-major, K contiguous. Nb % 64 == 0, K % 64 == 0, M % 16 == 0.
// One wave computes a 16x64 strip (4 WMMA tiles), software-pipelined one
// 32-wide K chunk deep so global_load_b128s overlap v_wmma issue.
// acc != 0 : accumulate onto existing C (fused x+s+h@W sums).
// C16      : optional f16 mirror of the output (feeds the next GEMM's A).
// NOTE: reads up to 32 halves past the end of A/B rows on the final pipeline
// stage; all workspace allocations carry >=256B slack to make this safe.
// ---------------------------------------------------------------------------
__global__ __launch_bounds__(GWAVES * 32)
void k_gemm(const h16* __restrict__ A, const h16* __restrict__ Bm,
            float* __restrict__ C, h16* __restrict__ C16,
            int M, int Nb, int K, int ldc, int Nstore,
            const float* __restrict__ bias, int acc)
{
    const int wave = threadIdx.x >> 5;
    const int lane = threadIdx.x & 31;
    const int hs   = lane >> 4;     // K sub-chunk select (half-wave)
    const int lr   = lane & 15;     // row within A tile / row of B (= output col)

    const int ngroups = Nb >> 6;                     // strips of 4 tiles
    const int gt = blockIdx.x * GWAVES + wave;
    const int tm = gt / ngroups;
    if (tm * 16 >= M) return;                        // wave-uniform exit
    const int tn0 = (gt % ngroups) * GNT;

    const h16* a_row = A + (size_t)(tm * 16 + lr) * K + hs * 8;
    const h16* b_row[GNT];
#pragma unroll
    for (int j = 0; j < GNT; ++j)
        b_row[j] = Bm + (size_t)((tn0 + j) * 16 + lr) * K + hs * 8;

    v8f acv[GNT];
    if (acc) {
#pragma unroll
        for (int j = 0; j < GNT; ++j) {
            const int ncol = (tn0 + j) * 16 + lr;
#pragma unroll
            for (int r = 0; r < 8; ++r)
                acv[j][r] = C[(size_t)(tm * 16 + r + 8 * hs) * ldc + ncol];
        }
    } else {
#pragma unroll
        for (int j = 0; j < GNT; ++j) {
            v8f z = {0.f, 0.f, 0.f, 0.f, 0.f, 0.f, 0.f, 0.f};
            acv[j] = z;
        }
    }

    auto ld16 = [](const h16* p) -> v16h {
        v8h lo = *(const v8h*)p;
        v8h hi = *(const v8h*)(p + 16);
        return __builtin_shufflevector(lo, hi,
                                       0,1,2,3,4,5,6,7,8,9,10,11,12,13,14,15);
    };

    // prologue: chunk 0 in flight
    v16h a0 = ld16(a_row);
    v16h b0[GNT];
#pragma unroll
    for (int j = 0; j < GNT; ++j) b0[j] = ld16(b_row[j]);

    for (int k = 0; k < K; k += 64) {
        // issue chunk k+32 while consuming chunk k
        v16h a1 = ld16(a_row + k + 32);
        v16h b1[GNT];
#pragma unroll
        for (int j = 0; j < GNT; ++j) b1[j] = ld16(b_row[j] + k + 32);
#pragma unroll
        for (int j = 0; j < GNT; ++j)
            acv[j] = __builtin_amdgcn_wmma_f32_16x16x32_f16(
                false, a0, false, b0[j], (short)0, acv[j], false, false);
        // issue chunk k+64 (harmless slack-padded over-read on final iter)
        a0 = ld16(a_row + k + 64);
#pragma unroll
        for (int j = 0; j < GNT; ++j) b0[j] = ld16(b_row[j] + k + 64);
#pragma unroll
        for (int j = 0; j < GNT; ++j)
            acv[j] = __builtin_amdgcn_wmma_f32_16x16x32_f16(
                false, a1, false, b1[j], (short)0, acv[j], false, false);
    }

#pragma unroll
    for (int j = 0; j < GNT; ++j) {
        const int ncol = (tn0 + j) * 16 + lr;
        if (ncol < Nstore) {                          // lane store predicate only
            const float bb = bias ? bias[ncol] : 0.0f;
#pragma unroll
            for (int r = 0; r < 8; ++r) {
                size_t off = (size_t)(tm * 16 + r + 8 * hs) * ldc + ncol;
                float v = acv[j][r] + bb;
                C[off] = v;
                if (C16) C16[off] = (h16)v;
            }
        }
    }
}

// ---------------------------------------------------------------------------
// Elementwise / data-movement kernels
// ---------------------------------------------------------------------------
__global__ void k_embed(const int* __restrict__ tok, const float* __restrict__ encW,
                        h16* __restrict__ emb)
{
    size_t idx = (size_t)blockIdx.x * blockDim.x + threadIdx.x;
    if (idx >= (size_t)MM * NHID) return;
    int row = (int)(idx >> 10), col = (int)(idx & 1023);
    emb[idx] = (h16)encW[(size_t)tok[row] * NHID + col];
}

__global__ void k_cvt_pad(const float* __restrict__ src, h16* __restrict__ dst,
                          int sr, int sc, int dr, int dc)
{
    size_t idx = (size_t)blockIdx.x * blockDim.x + threadIdx.x;
    size_t total = (size_t)dr * dc;
    if (idx >= total) return;
    int r = (int)(idx / dc), c = (int)(idx % dc);
    float v = (r < sr && c < sc) ? src[(size_t)r * sc + c] : 0.0f;
    dst[idx] = (h16)v;
}

// dst[r,c] = src[c,r] (src is [sr,sc]); pads with zero
__global__ void k_cvt_padT(const float* __restrict__ src, h16* __restrict__ dst,
                           int sr, int sc, int dr, int dc)
{
    size_t idx = (size_t)blockIdx.x * blockDim.x + threadIdx.x;
    size_t total = (size_t)dr * dc;
    if (idx >= total) return;
    int r = (int)(idx / dc), c = (int)(idx % dc);
    float v = (c < sr && r < sc) ? src[(size_t)c * sc + r] : 0.0f;
    dst[idx] = (h16)v;
}

__global__ void k_vadd3(const float* __restrict__ a, const float* __restrict__ b,
                        const float* __restrict__ c, float* __restrict__ dst, int n)
{
    int i = blockIdx.x * blockDim.x + threadIdx.x;
    if (i >= n) return;
    dst[i] = a[i] + (b ? b[i] : 0.0f) + (c ? c[i] : 0.0f);
}

// sememe_c = sg(si)*th(su);  sememe_h = sg(so)*th(sememe_c)
__global__ void k_sem_act(const float* __restrict__ ious, float* __restrict__ semc,
                          h16* __restrict__ semh)
{
    size_t idx = (size_t)blockIdx.x * blockDim.x + threadIdx.x;
    if (idx >= (size_t)MM * NHID) return;
    int m = (int)(idx >> 10), n = (int)(idx & 1023);
    const float* r = ious + (size_t)m * G3;
    float sc = sgm(r[n]) * tanhf(r[n + 2 * NHID]);
    semc[idx] = sc;
    semh[idx] = (h16)(sgm(r[n + NHID]) * tanhf(sc));
}

// sememe-gated LSTM cell (one timestep slice, 16x1024)
__global__ void k_cell(const float* __restrict__ iou, const float* __restrict__ xf,
                       const float* __restrict__ xfs, const float* __restrict__ sc,
                       float* __restrict__ cst, h16* __restrict__ hout)
{
    int idx = blockIdx.x * blockDim.x + threadIdx.x;
    if (idx >= BB * NHID) return;
    int m = idx >> 10, n = idx & 1023;
    const float* r = iou + (size_t)m * G3;
    float i = r[n], o = r[n + NHID], u = r[n + 2 * NHID];
    float c  = cst[idx];
    float cn = sgm(i) * tanhf(u) + sgm(xf[idx]) * c + sgm(xfs[idx]) * sc[idx];
    float hn = sgm(o) * tanhf(cn);
    cst[idx]  = cn;
    hout[idx] = (h16)hn;
}

// standard LSTM cell (PyTorch gate order i,f,g,o), one timestep slice
__global__ void k_lstm(const float* __restrict__ g4, float* __restrict__ cst,
                       float* __restrict__ hst, h16* __restrict__ hout)
{
    int idx = blockIdx.x * blockDim.x + threadIdx.x;
    if (idx >= BB * NHID) return;
    int m = idx >> 10, n = idx & 1023;
    const float* r = g4 + (size_t)m * G4;
    float i = r[n], f = r[n + NHID], gg = r[n + 2 * NHID], o = r[n + 3 * NHID];
    float c  = cst[idx];
    float cn = sgm(f) * c + sgm(i) * tanhf(gg);
    float hn = sgm(o) * tanhf(cn);
    cst[idx]  = cn;
    hst[idx]  = hn;
    hout[idx] = (h16)hn;
}

__global__ void k_finalize(const float* __restrict__ hst, const float* __restrict__ cst,
                           float* __restrict__ out)
{
    int idx = blockIdx.x * blockDim.x + threadIdx.x;
    if (idx >= BB * NHID) return;
    out[DEC_ELEMS + idx]             = hst[idx];
    out[DEC_ELEMS + BB * NHID + idx] = cst[idx];
}

// ---------------------------------------------------------------------------
// Host side
// ---------------------------------------------------------------------------
static void gemm(hipStream_t s, const h16* A, const h16* B, float* C, h16* C16,
                 int M, int Nb, int K, int ldc, int Nstore, const float* bias, int acc)
{
    int ngroups = Nb / 64;
    int totalw  = (M / 16) * ngroups;
    int blocks  = (totalw + GWAVES - 1) / GWAVES;
    k_gemm<<<blocks, GWAVES * 32, 0, s>>>(A, B, C, C16, M, Nb, K, ldc, Nstore, bias, acc);
}

static inline int ewb(size_t n) { return (int)((n + 255) / 256); }

extern "C" void kernel_launch(void* const* d_in, const int* in_sizes, int n_in,
                              void* d_out, int out_size, void* d_ws, size_t ws_size,
                              hipStream_t stream)
{
    (void)in_sizes; (void)n_in; (void)out_size; (void)ws_size;
    const int*   tokens = (const int*)  d_in[0];
    const float* inp_s  = (const float*)d_in[1];
    const float* h0     = (const float*)d_in[2];
    const float* c0     = (const float*)d_in[3];
    const float* enc_W  = (const float*)d_in[4];
    const float* sem_W  = (const float*)d_in[5];
    const float* ss_W   = (const float*)d_in[6];
    const float* ss_b   = (const float*)d_in[7];
    const float* ioux_W = (const float*)d_in[8];
    const float* ioux_b = (const float*)d_in[9];
    const float* iouh_W = (const float*)d_in[10];
    const float* iouh_b = (const float*)d_in[11];
    const float* ious_W = (const float*)d_in[12];
    const float* ious_b = (const float*)d_in[13];
    const float* fx_W   = (const float*)d_in[14];
    const float* fx_b   = (const float*)d_in[15];
    const float* fxs_W  = (const float*)d_in[16];
    const float* fxs_b  = (const float*)d_in[17];
    const float* fh_W   = (const float*)d_in[18];
    const float* fh_b   = (const float*)d_in[19];
    const float* fs_W   = (const float*)d_in[20];
    const float* fs_b   = (const float*)d_in[21];
    const float* Wih    = (const float*)d_in[22];
    const float* Whh    = (const float*)d_in[23];
    const float* bih    = (const float*)d_in[24];
    const float* bhh    = (const float*)d_in[25];
    const float* dec_W  = (const float*)d_in[26];
    const float* dec_b  = (const float*)d_in[27];
    float* out = (float*)d_out;

    // ---- workspace bump allocator; +256B slack per buffer for the GEMM's
    //      one-chunk-deep prefetch over-read ----
    char* w = (char*)d_ws;
    auto alloc = [&](size_t elems, size_t esz) -> void* {
        void* p = (void*)w;
        w += ((elems * esz + 255) & ~(size_t)255) + 256;
        return p;
    };
    h16*   inp16    = (h16*)  alloc((size_t)MM * SSP, 2);
    h16*   semWt16  = (h16*)  alloc((size_t)SDP * SSP, 2);
    float* semin32  = (float*)alloc((size_t)MM * SDP, 4);
    h16*   semin16  = (h16*)  alloc((size_t)MM * SDP, 2);
    h16*   ssW16    = (h16*)  alloc((size_t)G3 * SDP, 2);
    h16*   emb16    = (h16*)  alloc((size_t)MM * NHID, 2);
    h16*   iouxW16  = (h16*)  alloc((size_t)G3 * NHID, 2);
    h16*   iouhW16  = (h16*)  alloc((size_t)G3 * NHID, 2);
    h16*   iousW16  = (h16*)  alloc((size_t)G3 * NHID, 2);
    h16*   fxW16    = (h16*)  alloc((size_t)NHID * NHID, 2);
    h16*   fxsW16   = (h16*)  alloc((size_t)NHID * NHID, 2);
    h16*   fhW16    = (h16*)  alloc((size_t)NHID * NHID, 2);
    h16*   fsW16    = (h16*)  alloc((size_t)NHID * NHID, 2);
    h16*   WihW16   = (h16*)  alloc((size_t)G4 * NHID, 2);
    h16*   WhhW16   = (h16*)  alloc((size_t)G4 * NHID, 2);
    h16*   decW16   = (h16*)  alloc((size_t)VP * NHID, 2);
    float* ious32   = (float*)alloc((size_t)MM * G3, 4);
    float* semc32   = (float*)alloc((size_t)MM * NHID, 4);
    h16*   semh16   = (h16*)  alloc((size_t)MM * NHID, 2);
    float* xsiou32  = (float*)alloc((size_t)MM * G3, 4);
    float* xf32     = (float*)alloc((size_t)MM * NHID, 4);
    float* xfs32    = (float*)alloc((size_t)MM * NHID, 4);
    h16*   newin16  = (h16*)  alloc((size_t)MM * NHID, 2);
    float* xg32     = (float*)alloc((size_t)MM * G4, 4);
    h16*   h0_16    = (h16*)  alloc((size_t)BB * NHID, 2);
    h16*   lstmh16  = (h16*)  alloc((size_t)MM * NHID, 2);
    float* cstate   = (float*)alloc((size_t)BB * NHID, 4);
    float* cl32     = (float*)alloc((size_t)BB * NHID, 4);
    float* h32      = (float*)alloc((size_t)BB * NHID, 4);
    float* b_iou    = (float*)alloc(G3, 4);
    float* b_f      = (float*)alloc(NHID, 4);
    float* b_fs     = (float*)alloc(NHID, 4);
    float* b_lstm   = (float*)alloc(G4, 4);

    // ---- converts / pads / gathers ----
    k_cvt_pad <<<ewb((size_t)MM * SSP), 256, 0, stream>>>(inp_s, inp16, MM, SS, MM, SSP);
    k_cvt_padT<<<ewb((size_t)SDP * SSP), 256, 0, stream>>>(sem_W, semWt16, SS, SD, SDP, SSP);
    k_cvt_pad <<<ewb((size_t)G3 * SDP), 256, 0, stream>>>(ss_W, ssW16, G3, SD, G3, SDP);
    k_embed   <<<ewb((size_t)MM * NHID), 256, 0, stream>>>(tokens, enc_W, emb16);
    k_cvt_pad <<<ewb((size_t)G3 * NHID), 256, 0, stream>>>(ioux_W, iouxW16, G3, NHID, G3, NHID);
    k_cvt_pad <<<ewb((size_t)G3 * NHID), 256, 0, stream>>>(iouh_W, iouhW16, G3, NHID, G3, NHID);
    k_cvt_pad <<<ewb((size_t)G3 * NHID), 256, 0, stream>>>(ious_W, iousW16, G3, NHID, G3, NHID);
    k_cvt_pad <<<ewb((size_t)NHID * NHID), 256, 0, stream>>>(fx_W,  fxW16,  NHID, NHID, NHID, NHID);
    k_cvt_pad <<<ewb((size_t)NHID * NHID), 256, 0, stream>>>(fxs_W, fxsW16, NHID, NHID, NHID, NHID);
    k_cvt_pad <<<ewb((size_t)NHID * NHID), 256, 0, stream>>>(fh_W,  fhW16,  NHID, NHID, NHID, NHID);
    k_cvt_pad <<<ewb((size_t)NHID * NHID), 256, 0, stream>>>(fs_W,  fsW16,  NHID, NHID, NHID, NHID);
    k_cvt_pad <<<ewb((size_t)G4 * NHID), 256, 0, stream>>>(Wih, WihW16, G4, NHID, G4, NHID);
    k_cvt_pad <<<ewb((size_t)G4 * NHID), 256, 0, stream>>>(Whh, WhhW16, G4, NHID, G4, NHID);
    k_cvt_pad <<<ewb((size_t)VP * NHID), 256, 0, stream>>>(dec_W, decW16, VV, NHID, VP, NHID);
    k_cvt_pad <<<ewb((size_t)BB * NHID), 256, 0, stream>>>(h0, h0_16, BB, NHID, BB, NHID);
    k_vadd3<<<ewb(G3), 256, 0, stream>>>(ioux_b, iouh_b, ious_b, b_iou, G3);
    k_vadd3<<<ewb(NHID), 256, 0, stream>>>(fx_b,  fh_b, nullptr, b_f,  NHID);
    k_vadd3<<<ewb(NHID), 256, 0, stream>>>(fxs_b, fs_b, nullptr, b_fs, NHID);
    k_vadd3<<<ewb(G4), 256, 0, stream>>>(bih, bhh, nullptr, b_lstm, G4);
    hipMemsetAsync(cstate, 0, (size_t)BB * NHID * 4, stream);
    hipMemcpyAsync(cl32, c0, (size_t)BB * NHID * 4, hipMemcpyDeviceToDevice, stream);

    // ---- sememe path: sem_in = input_s @ sem_W ; iou_s = sem_in @ ss_W^T + ss_b ----
    gemm(stream, inp16, semWt16, semin32, semin16, MM, SDP, SSP, SDP, SDP, nullptr, 0);
    gemm(stream, semin16, ssW16, ious32, nullptr, MM, G3, SDP, G3, G3, ss_b, 0);
    k_sem_act<<<ewb((size_t)MM * NHID), 256, 0, stream>>>(ious32, semc32, semh16);

    // ---- precomputed projections (fused accumulation into shared buffers) ----
    gemm(stream, emb16,  iouxW16, xsiou32, nullptr, MM, G3, NHID, G3, G3, b_iou, 0);
    gemm(stream, semh16, iousW16, xsiou32, nullptr, MM, G3, NHID, G3, G3, nullptr, 1);
    gemm(stream, emb16,  fxW16,   xf32,    nullptr, MM, NHID, NHID, NHID, NHID, b_f,  0);
    gemm(stream, emb16,  fxsW16,  xfs32,   nullptr, MM, NHID, NHID, NHID, NHID, b_fs, 0);
    gemm(stream, semh16, fsW16,   xfs32,   nullptr, MM, NHID, NHID, NHID, NHID, nullptr, 1);

    // ---- sememe-gated LSTM scan (h_{t-1} @ W accumulated into slice t) ----
    for (int t = 0; t < TT; ++t) {
        if (t > 0) {
            const h16* hp = newin16 + (size_t)(t - 1) * BB * NHID;
            gemm(stream, hp, iouhW16, xsiou32 + (size_t)t * BB * G3,  nullptr,
                 BB, G3, NHID, G3, G3, nullptr, 1);
            gemm(stream, hp, fhW16,   xf32    + (size_t)t * BB * NHID, nullptr,
                 BB, NHID, NHID, NHID, NHID, nullptr, 1);
        }
        k_cell<<<ewb((size_t)BB * NHID), 256, 0, stream>>>(
            xsiou32 + (size_t)t * BB * G3,
            xf32    + (size_t)t * BB * NHID,
            xfs32   + (size_t)t * BB * NHID,
            semc32  + (size_t)t * BB * NHID,
            cstate,
            newin16 + (size_t)t * BB * NHID);
    }

    // ---- standard LSTM: x_gates then scan ----
    gemm(stream, newin16, WihW16, xg32, nullptr, MM, G4, NHID, G4, G4, b_lstm, 0);
    for (int t = 0; t < TT; ++t) {
        const h16* hp = (t == 0) ? h0_16 : (lstmh16 + (size_t)(t - 1) * BB * NHID);
        gemm(stream, hp, WhhW16, xg32 + (size_t)t * BB * G4, nullptr,
             BB, G4, NHID, G4, G4, nullptr, 1);
        k_lstm<<<ewb((size_t)BB * NHID), 256, 0, stream>>>(
            xg32 + (size_t)t * BB * G4, cl32, h32,
            lstmh16 + (size_t)t * BB * NHID);
    }

    // ---- decoder: output @ dec_W^T + dec_b  (dec_W f16 = 103 MB, L2-resident) ----
    gemm(stream, lstmh16, decW16, out, nullptr, MM, VP, NHID, VV, VV, dec_b, 0);

    // ---- hT, cT tail of d_out ----
    k_finalize<<<ewb((size_t)BB * NHID), 256, 0, stream>>>(h32, cl32, out);
}
